// PCNet_47296179863787
// MI455X (gfx1250) — matrix operand order repaired and checked by reading, not compiled, not run on hardware
//
#include <hip/hip_runtime.h>
#include <hip/hip_bf16.h>

// ---------------------------------------------------------------------------
// PCNet (ISTA sparse coding) on MI455X / gfx1250.
// All GEMMs run on the bf16 WMMA pipe (v_wmma_f32_16x16x32_bf16) with f32
// accumulation; R state kept in f32. Whole working set (~20 MB) is L2-resident
// (192 MB), so fragments stream directly from L2 (no LDS staging).
// ---------------------------------------------------------------------------

typedef __attribute__((ext_vector_type(16))) __bf16 v16bf;
typedef __attribute__((ext_vector_type(8)))  float  v8f;

#define MDIM 2048   // batch
#define NDIM 1024   // output cols (neurons or pixels; both are 1024)
#define KDIM 1024   // reduction dim (both GEMMs reduce over 1024)

#define R_LR   0.1f
#define LMDA   0.005f
#define EPOCHS 150

__device__ __forceinline__ unsigned short f32_to_bf16(float f) {
  unsigned int u = __float_as_uint(f);
  u += 0x7FFFu + ((u >> 16) & 1u);          // round-to-nearest-even
  return (unsigned short)(u >> 16);
}

union Frag { uint4 u[2]; v16bf v; };        // 32 bytes = 16 bf16 = 8 VGPRs

// --- one-time prep: bf16 copies of U and U^T -------------------------------
__global__ void prep_U(const float* __restrict__ U,
                       unsigned short* __restrict__ Ub,
                       unsigned short* __restrict__ Utb) {
  int idx = blockIdx.x * blockDim.x + threadIdx.x;   // over 1024*1024
  int k = idx >> 10;          // row of U (neuron)
  int m = idx & 1023;         // col of U (pixel)
  unsigned short h = f32_to_bf16(U[idx]);
  Ub[idx] = h;                // Ub[k][m]
  Utb[m * KDIM + k] = h;      // Utb[m][k] = U[k][m]
}

__global__ void zero_R(float* __restrict__ Rf, unsigned short* __restrict__ Rb) {
  int idx = blockIdx.x * blockDim.x + threadIdx.x;
  Rf[idx] = 0.0f;
  Rb[idx] = 0;
}

// --- fused GEMM (+ epilogue) -----------------------------------------------
// Computes D = A(MxK bf16) * B(KxN bf16), where B is supplied transposed as
// BT (NxK row-major) so every lane's B-fragment is a contiguous 32B load.
// MODE 0: Eb = bf16(D - img)                       (A=Rb, BT=U^T)
// MODE 1: R  = soft(R - 2*lr*D, lmda); Rb = bf16(R) (A=Eb, BT=U)
// MODE 2: Out = D (f32)                            (A=Rb, BT=U^T)
template <int MODE>
__global__ __launch_bounds__(256)
void pc_gemm(const unsigned short* __restrict__ A,
             const unsigned short* __restrict__ BT,
             const float* __restrict__ img,
             unsigned short* __restrict__ Eb,
             float* __restrict__ Rf,
             unsigned short* __restrict__ Rb,
             float* __restrict__ Out) {
  const int lane = threadIdx.x & 31;
  const int wave = threadIdx.x >> 5;      // 0..7
  const int wm   = wave >> 2;             // 0..1  (M direction)
  const int wn   = wave & 3;              // 0..3  (N direction)
  const int l15  = lane & 15;
  const int lhi  = lane >> 4;             // 0/1 (half-wave)

  const int bm = blockIdx.x * 128;
  const int bn = blockIdx.y * 128;

  // accumulators: 4 M-tiles x 2 N-tiles of 16x16 f32
  v8f acc[4][2];
#pragma unroll
  for (int i = 0; i < 4; ++i)
#pragma unroll
    for (int j = 0; j < 2; ++j) acc[i][j] = (v8f){};

  // per-lane row base pointers (contiguous-K gathers, same k order for A and B)
  const unsigned short* arow[4];
#pragma unroll
  for (int im = 0; im < 4; ++im)
    arow[im] = A + (size_t)(bm + wm * 64 + im * 16 + l15) * KDIM;
  const unsigned short* brow[2];
#pragma unroll
  for (int in = 0; in < 2; ++in)
    brow[in] = BT + (size_t)(bn + wn * 32 + in * 16 + l15) * KDIM;

  for (int k0 = 0; k0 < KDIM; k0 += 32) {
    Frag a[4], b[2];
#pragma unroll
    for (int im = 0; im < 4; ++im) {
      const uint4* p = (const uint4*)(arow[im] + k0);
      a[im].u[0] = p[2 * lhi];        // K = k0 + 16*lhi .. +15 (contig 32B)
      a[im].u[1] = p[2 * lhi + 1];
    }
#pragma unroll
    for (int in = 0; in < 2; ++in) {
      const uint4* p = (const uint4*)(brow[in] + k0);
      b[in].u[0] = p[2 * lhi];        // K = k0 + 16*lhi .. +15 (contig 32B)
      b[in].u[1] = p[2 * lhi + 1];
    }
#pragma unroll
    for (int im = 0; im < 4; ++im)
#pragma unroll
      for (int in = 0; in < 2; ++in)
        acc[im][in] = __builtin_amdgcn_wmma_f32_16x16x32_bf16(
            false, a[im].v, false, b[in].v, (short)0, acc[im][in],
            false, false);
  }

  // epilogue: D element r of lane -> (m = r + 8*lhi, n = l15) within tile
#pragma unroll
  for (int im = 0; im < 4; ++im) {
#pragma unroll
    for (int in = 0; in < 2; ++in) {
      const int n = bn + wn * 32 + in * 16 + l15;
#pragma unroll
      for (int r = 0; r < 8; ++r) {
        const int m = bm + wm * 64 + im * 16 + r + 8 * lhi;
        const size_t idx = (size_t)m * NDIM + n;
        const float v = acc[im][in][r];
        if (MODE == 0) {
          Eb[idx] = f32_to_bf16(v - img[idx]);
        } else if (MODE == 1) {
          float x  = Rf[idx] - (2.0f * R_LR) * v;
          float ax = fabsf(x) - LMDA;
          float s  = ax > 0.0f ? (x > 0.0f ? ax : -ax) : 0.0f;
          Rf[idx] = s;
          Rb[idx] = f32_to_bf16(s);
        } else {
          Out[idx] = v;
        }
      }
    }
  }
}

extern "C" void kernel_launch(void* const* d_in, const int* in_sizes, int n_in,
                              void* d_out, int out_size, void* d_ws, size_t ws_size,
                              hipStream_t stream) {
  const float* img = (const float*)d_in[0];   // (2048, 1024) f32
  const float* U   = (const float*)d_in[1];   // (1024, 1024) f32
  float* out = (float*)d_out;                 // (2048, 1024) f32

  // workspace layout (20 MB total)
  char* ws = (char*)d_ws;
  unsigned short* Ub  = (unsigned short*)(ws);                         // 2 MB
  unsigned short* Utb = (unsigned short*)(ws + (size_t)(2u << 20));    // 2 MB
  unsigned short* Rb  = (unsigned short*)(ws + (size_t)(4u << 20));    // 4 MB
  unsigned short* Eb  = (unsigned short*)(ws + (size_t)(8u << 20));    // 4 MB
  float*          Rf  = (float*)        (ws + (size_t)(12u << 20));    // 8 MB

  prep_U<<<(NDIM * KDIM) / 256, 256, 0, stream>>>(U, Ub, Utb);
  zero_R<<<(MDIM * KDIM) / 256, 256, 0, stream>>>(Rf, Rb);

  dim3 grid(MDIM / 128, NDIM / 128);   // 16 x 8 workgroups
  for (int e = 0; e < EPOCHS; ++e) {
    // E = Rb @ U - img      (B of this GEMM is U, supplied transposed = Utb)
    pc_gemm<0><<<grid, 256, 0, stream>>>(Rb, Utb, img, Eb,
                                         nullptr, nullptr, nullptr);
    // G = Eb @ U^T ; R update  (B is U^T, supplied transposed = Ub)
    pc_gemm<1><<<grid, 256, 0, stream>>>(Eb, Ub, nullptr, nullptr,
                                         Rf, Rb, nullptr);
  }
  // out = Rb @ U
  pc_gemm<2><<<grid, 256, 0, stream>>>(Rb, Utb, nullptr, nullptr,
                                       nullptr, nullptr, out);
}
